// DotGATLayer_19542101196804
// MI455X (gfx1250) — matrix operand order repaired and checked by reading, not compiled, not run on hardware
//
#include <hip/hip_runtime.h>
#include <cmath>

typedef __attribute__((ext_vector_type(16))) _Float16 v16h;
typedef __attribute__((ext_vector_type(8)))  _Float16 v8h;
typedef __attribute__((ext_vector_type(8)))  float    v8f;

#define BATCH 4
#define SEQ   4096
#define DIM   128
#define BA    (BATCH * SEQ)

#define LOG2E 1.4426950408889634f

// ---------- wave32 half-warp (16-lane) reductions: stay within C-tile halves ----------
__device__ __forceinline__ float half_red_max(float v) {
    v = fmaxf(v, __shfl_xor(v, 1, 32));
    v = fmaxf(v, __shfl_xor(v, 2, 32));
    v = fmaxf(v, __shfl_xor(v, 4, 32));
    v = fmaxf(v, __shfl_xor(v, 8, 32));
    return v;
}
__device__ __forceinline__ float half_red_sum(float v) {
    v += __shfl_xor(v, 1, 32);
    v += __shfl_xor(v, 2, 32);
    v += __shfl_xor(v, 4, 32);
    v += __shfl_xor(v, 8, 32);
    return v;
}
__device__ __forceinline__ float swishf(float x) {
    return x / (1.f + __expf(-x));
}

// ---------- kernel 0: f32 [din][dout] -> f16 [dout][din] for WMMA B-fragments ----------
__global__ void prep_weights(const float* __restrict__ Wq, const float* __restrict__ Wk,
                             const float* __restrict__ Wv, const float* __restrict__ W1,
                             const float* __restrict__ W2,
                             _Float16* __restrict__ Wqt, _Float16* __restrict__ Wkt,
                             _Float16* __restrict__ Wvt, _Float16* __restrict__ W1t,
                             _Float16* __restrict__ W2t) {
    const float* src = (blockIdx.x == 0) ? Wq : (blockIdx.x == 1) ? Wk :
                       (blockIdx.x == 2) ? Wv : (blockIdx.x == 3) ? W1 : W2;
    _Float16* dst = (blockIdx.x == 0) ? Wqt : (blockIdx.x == 1) ? Wkt :
                    (blockIdx.x == 2) ? Wvt : (blockIdx.x == 3) ? W1t : W2t;
    for (int i = threadIdx.x; i < DIM * DIM; i += blockDim.x) {
        int din = i >> 7, dout = i & 127;
        dst[dout * DIM + din] = (_Float16)src[i];
    }
}

// ---------- kernel 1: QKV projections; Q pre-scaled by log2e/sqrt(128); V stored transposed ----------
__global__ __launch_bounds__(128) void qkv_kernel(
        const float* __restrict__ x,
        const _Float16* __restrict__ Wqt, const _Float16* __restrict__ Wkt,
        const _Float16* __restrict__ Wvt,
        _Float16* __restrict__ Qs, _Float16* __restrict__ Kh, _Float16* __restrict__ Vt) {
    const int wave = threadIdx.x >> 5, lane = threadIdx.x & 31;
    const int lo = lane & 15, hi = lane >> 4;
    const int row0 = (blockIdx.x * 4 + wave) * 16;
    const int bb = row0 / SEQ, a0 = row0 % SEQ;

    // A fragments of x (f32 -> f16), contiguous k order, same as B: row=lane%16, K=16hi..16hi+15
    v16h xa[4];
    const float* xrow = x + (size_t)(row0 + lo) * DIM;
    #pragma unroll
    for (int c = 0; c < 4; ++c) {
        union { v16h v; _Float16 e[16]; } u;
        const float* p0 = xrow + c * 32 + 16 * hi;
        const float* p1 = xrow + c * 32 + 16 * hi + 8;
        #pragma unroll
        for (int j = 0; j < 8; ++j) { u.e[j] = (_Float16)p0[j]; u.e[8 + j] = (_Float16)p1[j]; }
        xa[c] = u.v;
    }

    const float qscale = LOG2E / 11.313708498984761f; // log2e / sqrt(128): base-2 softmax
    #pragma unroll 1
    for (int n = 0; n < 8; ++n) {
        v8f qacc = {}, kacc = {}, vacc = {};
        const int colbase = (n * 16 + lo) * DIM + 16 * hi;
        #pragma unroll
        for (int c = 0; c < 4; ++c) {
            v16h bq = *(const v16h*)(Wqt + colbase + c * 32);
            qacc = __builtin_amdgcn_wmma_f32_16x16x32_f16(false, xa[c], false, bq, (short)0, qacc, false, false);
            v16h bk = *(const v16h*)(Wkt + colbase + c * 32);
            kacc = __builtin_amdgcn_wmma_f32_16x16x32_f16(false, xa[c], false, bk, (short)0, kacc, false, false);
            v16h bv = *(const v16h*)(Wvt + colbase + c * 32);
            vacc = __builtin_amdgcn_wmma_f32_16x16x32_f16(false, xa[c], false, bv, (short)0, vacc, false, false);
        }
        const int dcol = n * 16 + lo;
        #pragma unroll
        for (int r = 0; r < 8; ++r) {
            const int row = row0 + r + 8 * hi;
            Qs[(size_t)row * DIM + dcol] = (_Float16)(qacc[r] * qscale);
            Kh[(size_t)row * DIM + dcol] = (_Float16)kacc[r];
            const int a = a0 + r + 8 * hi;
            Vt[((size_t)bb * DIM + dcol) * SEQ + a] = (_Float16)vacc[r];
        }
    }
}

// ---------- kernel 2: fused flash attention, 64-key blocks, base-2 softmax ----------
__global__ __launch_bounds__(128) void attn_kernel(
        const float* __restrict__ conn,
        const _Float16* __restrict__ Qs, const _Float16* __restrict__ Kh,
        const _Float16* __restrict__ Vt, float* __restrict__ H) {
    __shared__ _Float16 plds[4 * 16 * 64];   // per-wave P staging (C-layout -> A-layout)
    const int wave = threadIdx.x >> 5, lane = threadIdx.x & 31;
    const int lo = lane & 15, hi = lane >> 4;
    const int row0 = (blockIdx.x * 4 + wave) * 16;
    const int bb = row0 / SEQ, q0 = row0 % SEQ;
    _Float16* pbase = plds + wave * (16 * 64);

    // Q A-fragments (pre-scaled by log2e/sqrt(128))
    v16h qa[4];
    const _Float16* qrow = Qs + (size_t)(row0 + lo) * DIM;
    #pragma unroll
    for (int c = 0; c < 4; ++c) {
        union { v16h v; v8h h[2]; } u;
        u.h[0] = *(const v8h*)(qrow + c * 32 + 16 * hi);
        u.h[1] = *(const v8h*)(qrow + c * 32 + 16 * hi + 8);
        qa[c] = u.v;
    }
    // all-ones B fragment: row-sums of P via the matrix core (replaces shfl reductions)
    v16h ones;
    {
        union { v16h v; _Float16 e[16]; } u;
        #pragma unroll
        for (int j = 0; j < 16; ++j) u.e[j] = (_Float16)1.0f;
        ones = u.v;
    }

    float m[8], lsum[8];
    v8f hacc[8];
    #pragma unroll
    for (int r = 0; r < 8; ++r) { m[r] = -INFINITY; lsum[r] = 0.f; }
    #pragma unroll
    for (int t = 0; t < 8; ++t) hacc[t] = (v8f){};

    const float* cbase = conn + (size_t)bb * SEQ * SEQ + (size_t)q0 * SEQ;
    const _Float16* kbbase = Kh + (size_t)bb * SEQ * DIM;
    const _Float16* vbbase = Vt + (size_t)bb * DIM * SEQ;

    for (int k0 = 0; k0 < SEQ; k0 += 64) {
        // connectivity bias (x log2e) preloaded into WMMA accumulators; streamed nontemporal
        v8f s[4];
        #pragma unroll
        for (int r = 0; r < 8; ++r) {
            const float* cp = cbase + (size_t)(r + 8 * hi) * SEQ + k0 + lo;
            #pragma unroll
            for (int nt = 0; nt < 4; ++nt)
                s[nt][r] = __builtin_nontemporal_load(cp + nt * 16) * LOG2E;
        }
        // S2 = (Q*log2e/sqrt(128)) * K^T + conn*log2e   (four 16-key n-tiles)
        #pragma unroll
        for (int nt = 0; nt < 4; ++nt) {
            const _Float16* krow = kbbase + (size_t)(k0 + nt * 16 + lo) * DIM + 16 * hi;
            #pragma unroll
            for (int c = 0; c < 4; ++c) {
                v16h b = *(const v16h*)(krow + c * 32);
                s[nt] = __builtin_amdgcn_wmma_f32_16x16x32_f16(false, qa[c], false, b, (short)0, s[nt], false, false);
            }
        }
        // online softmax (base 2) per row; row r+8*hi lives across the 16 lanes of this half
        #pragma unroll
        for (int r = 0; r < 8; ++r) {
            float rmax = fmaxf(fmaxf(s[0][r], s[1][r]), fmaxf(s[2][r], s[3][r]));
            rmax = half_red_max(rmax);
            const float mnew = fmaxf(m[r], rmax);
            const float sc = __builtin_amdgcn_exp2f(m[r] - mnew);
            m[r] = mnew;
            lsum[r] *= sc;
            #pragma unroll
            for (int t = 0; t < 8; ++t) hacc[t][r] *= sc;
            _Float16* pp = pbase + (r + 8 * hi) * 64 + lo;
            #pragma unroll
            for (int nt = 0; nt < 4; ++nt)
                pp[nt * 16] = (_Float16)__builtin_amdgcn_exp2f(s[nt][r] - mnew);
        }
        asm volatile("s_wait_dscnt 0" ::: "memory");   // cross-lane LDS handoff within wave
        // P as two 16x32 A-fragments
        union { v16h v; v8h h[2]; } pu;
        const _Float16* prow = pbase + lo * 64;
        pu.h[0] = *(const v8h*)(prow + 16 * hi);
        pu.h[1] = *(const v8h*)(prow + 16 * hi + 8);
        v16h pa0 = pu.v;
        pu.h[0] = *(const v8h*)(prow + 32 + 16 * hi);
        pu.h[1] = *(const v8h*)(prow + 40 + 16 * hi);
        v16h pa1 = pu.v;
        asm volatile("" ::: "memory");                 // keep next-iter stores after these loads
        // row-sums of P with the matrix core: rs[r] broadcast across the half
        v8f rs = {};
        rs = __builtin_amdgcn_wmma_f32_16x16x32_f16(false, pa0, false, ones, (short)0, rs, false, false);
        rs = __builtin_amdgcn_wmma_f32_16x16x32_f16(false, pa1, false, ones, (short)0, rs, false, false);
        #pragma unroll
        for (int r = 0; r < 8; ++r) lsum[r] += rs[r];
        // H += P * V  (V transposed: per-lane contiguous k-strips)
        const _Float16* vrow = vbbase + (size_t)lo * SEQ + k0 + 16 * hi;
        #pragma unroll
        for (int t = 0; t < 8; ++t) {
            const _Float16* vp = vrow + (size_t)t * 16 * SEQ;
            v16h vb0 = *(const v16h*)(vp);
            hacc[t] = __builtin_amdgcn_wmma_f32_16x16x32_f16(false, pa0, false, vb0, (short)0, hacc[t], false, false);
            v16h vb1 = *(const v16h*)(vp + 32);
            hacc[t] = __builtin_amdgcn_wmma_f32_16x16x32_f16(false, pa1, false, vb1, (short)0, hacc[t], false, false);
        }
    }
    // normalize (reciprocal instead of 64 divides) and write H
    float rl[8];
    #pragma unroll
    for (int r = 0; r < 8; ++r) rl[r] = 1.f / lsum[r];
    #pragma unroll
    for (int t = 0; t < 8; ++t) {
        const int d = t * 16 + lo;
        #pragma unroll
        for (int r = 0; r < 8; ++r)
            H[(size_t)(row0 + r + 8 * hi) * DIM + d] = hacc[t][r] * rl[r];
    }
}

// ---------- kernel 3: swish -> Linear(+b1) -> swish -> Linear(+b2) -> LayerNorm ----------
__global__ __launch_bounds__(128) void mlp_ln_kernel(
        const float* __restrict__ Hb,
        const _Float16* __restrict__ W1t, const _Float16* __restrict__ W2t,
        const float* __restrict__ b1, const float* __restrict__ b2,
        const float* __restrict__ gamma, const float* __restrict__ beta,
        float* __restrict__ out) {
    __shared__ _Float16 gl[4 * 16 * DIM];
    const int wave = threadIdx.x >> 5, lane = threadIdx.x & 31;
    const int lo = lane & 15, hi = lane >> 4;
    const int row0 = (blockIdx.x * 4 + wave) * 16;
    _Float16* gbase = gl + wave * (16 * DIM);

    // A fragments of swish(H)
    v16h ha[4];
    const float* hrow = Hb + (size_t)(row0 + lo) * DIM;
    #pragma unroll
    for (int c = 0; c < 4; ++c) {
        union { v16h v; _Float16 e[16]; } u;
        const float* p0 = hrow + c * 32 + 16 * hi;
        const float* p1 = hrow + c * 32 + 16 * hi + 8;
        #pragma unroll
        for (int j = 0; j < 8; ++j) {
            u.e[j]     = (_Float16)swishf(p0[j]);
            u.e[8 + j] = (_Float16)swishf(p1[j]);
        }
        ha[c] = u.v;
    }
    // GEMM1 + b1, swish, stage to LDS row-major
    #pragma unroll 1
    for (int n = 0; n < 8; ++n) {
        const float bv = b1[n * 16 + lo];
        v8f acc = { bv, bv, bv, bv, bv, bv, bv, bv };
        const _Float16* wcol = W1t + (n * 16 + lo) * DIM + 16 * hi;
        #pragma unroll
        for (int c = 0; c < 4; ++c) {
            v16h b = *(const v16h*)(wcol + c * 32);
            acc = __builtin_amdgcn_wmma_f32_16x16x32_f16(false, ha[c], false, b, (short)0, acc, false, false);
        }
        #pragma unroll
        for (int r = 0; r < 8; ++r)
            gbase[(r + 8 * hi) * DIM + n * 16 + lo] = (_Float16)swishf(acc[r]);
    }
    asm volatile("s_wait_dscnt 0" ::: "memory");
    v16h ga[4];
    #pragma unroll
    for (int c = 0; c < 4; ++c) {
        union { v16h v; v8h h[2]; } u;
        const _Float16* g = gbase + lo * DIM + c * 32;
        u.h[0] = *(const v8h*)(g + 16 * hi);
        u.h[1] = *(const v8h*)(g + 16 * hi + 8);
        ga[c] = u.v;
    }
    // GEMM2 + b2, keep all 8 d-tiles for LayerNorm
    v8f o[8];
    #pragma unroll 1
    for (int n = 0; n < 8; ++n) {
        const float bv = b2[n * 16 + lo];
        v8f acc = { bv, bv, bv, bv, bv, bv, bv, bv };
        const _Float16* wcol = W2t + (n * 16 + lo) * DIM + 16 * hi;
        #pragma unroll
        for (int c = 0; c < 4; ++c) {
            v16h b = *(const v16h*)(wcol + c * 32);
            acc = __builtin_amdgcn_wmma_f32_16x16x32_f16(false, ga[c], false, b, (short)0, acc, false, false);
        }
        o[n] = acc;
    }
    float gm[8], bt[8];
    #pragma unroll
    for (int t = 0; t < 8; ++t) { gm[t] = gamma[t * 16 + lo]; bt[t] = beta[t * 16 + lo]; }
    // LayerNorm over d=128 (8 tiles x 16 lanes)
    #pragma unroll
    for (int r = 0; r < 8; ++r) {
        float sacc = 0.f, s2 = 0.f;
        #pragma unroll
        for (int t = 0; t < 8; ++t) { const float v = o[t][r]; sacc += v; s2 += v * v; }
        sacc = half_red_sum(sacc);
        s2   = half_red_sum(s2);
        const float mu   = sacc * (1.f / 128.f);
        const float var  = s2 * (1.f / 128.f) - mu * mu;
        const float rstd = rsqrtf(var + 1e-5f);
        #pragma unroll
        for (int t = 0; t < 8; ++t) {
            const float v = (o[t][r] - mu) * rstd * gm[t] + bt[t];
            out[(size_t)(row0 + r + 8 * hi) * DIM + t * 16 + lo] = v;
        }
    }
}

extern "C" void kernel_launch(void* const* d_in, const int* in_sizes, int n_in,
                              void* d_out, int out_size, void* d_ws, size_t ws_size,
                              hipStream_t stream) {
    (void)in_sizes; (void)n_in; (void)out_size; (void)ws_size;
    const float* x    = (const float*)d_in[0];
    const float* conn = (const float*)d_in[1];
    const float* Wq   = (const float*)d_in[2];
    const float* Wk   = (const float*)d_in[3];
    const float* Wv   = (const float*)d_in[4];
    const float* W1   = (const float*)d_in[5];
    const float* b1   = (const float*)d_in[6];
    const float* W2   = (const float*)d_in[7];
    const float* b2   = (const float*)d_in[8];
    const float* gam  = (const float*)d_in[9];
    const float* bet  = (const float*)d_in[10];
    float* out = (float*)d_out;

    char* ws = (char*)d_ws;
    _Float16* Qs  = (_Float16*)(ws);                          // 4 MiB
    _Float16* Kh  = (_Float16*)(ws + ((size_t)4  << 20));     // 4 MiB
    _Float16* Vt  = (_Float16*)(ws + ((size_t)8  << 20));     // 4 MiB [b][d][a]
    float*    Hb  = (float*)   (ws + ((size_t)12 << 20));     // 8 MiB
    _Float16* Wqt = (_Float16*)(ws + ((size_t)20 << 20));     // 5 x 32 KiB
    _Float16* Wkt = Wqt + DIM * DIM;
    _Float16* Wvt = Wkt + DIM * DIM;
    _Float16* W1t = Wvt + DIM * DIM;
    _Float16* W2t = W1t + DIM * DIM;

    prep_weights<<<5, 256, 0, stream>>>(Wq, Wk, Wv, W1, W2, Wqt, Wkt, Wvt, W1t, W2t);
    qkv_kernel   <<<BA / 64, 128, 0, stream>>>(x, Wqt, Wkt, Wvt, Qs, Kh, Vt);
    attn_kernel  <<<BA / 64, 128, 0, stream>>>(conn, Qs, Kh, Vt, Hb);
    mlp_ln_kernel<<<BA / 64, 128, 0, stream>>>(Hb, W1t, W2t, b1, b2, gam, bet, out);
}
